// PointCnnLayer_13202729467907
// MI455X (gfx1250) — matrix operand order repaired and checked by reading, not compiled, not run on hardware
//
#include <hip/hip_runtime.h>
#include <cstddef>
#include <cstdint>

// =====================================================================
// PointCNN forward for MI455X (gfx1250, wave32, WMMA).
//
// Round-2 changes: all WMMA operands are pre-packed into zero-padded
// f16 buffers so the GEMM/fx inner loops contain ONLY unconditional
// loads (16-byte vectors for A, strided halves for B) + v_wmma (no predicated scalar loads, no
// per-element waits, EXEC untouched). Zero padding in the A columns
// beyond Kd makes out-of-range B rows contribute 0, so no guards are
// needed in the K-loop at all.
//
// The dwo/pw pair of each XConv is fused into Wf[k*Ccat+c,o] =
// sum_d dw[c,d,k]*pw[c*dm+d,o]; fx and wf_fuse emit f16 directly so the
// dominant projection GEMM needs no pack pass.
// =====================================================================

typedef __attribute__((ext_vector_type(16))) _Float16 v16h;
typedef __attribute__((ext_vector_type(8)))  _Float16 v8h;
typedef __attribute__((ext_vector_type(8)))  float    v8f;

union V16 { v16h v; v8h h[2]; };

#define TPB 256
#define MAX_DK1 65   // largest D*K+1 in the network (K=16, D=4)

__device__ __forceinline__ float elu1(float v) {
  return v > 0.f ? v : (__expf(v) - 1.f);
}

// ---------------------------------------------------------------------
// WMMA GEMM on packed f16: C[M,N] = act(A @ W + bias)
//   A: f16 [M, Ka] row-major, Ka = roundup32(Kd), cols >= Kd are zero
//   W: f16 [Ka, Np] row-major, Np = roundup32(N), pad rows/cols zero
// One wave computes a 16x32 tile (2 wmma per K-step, A loaded once).
// Fragment layouts per CDNA5 ISA 7.12.2 (wave32):
//   A lane m=l&15 row m; halves [k0+8hi .. +7] and [k0+16+8hi .. +7]
//   B lane l col l&15; halves [k0+8hi .. +7] and [k0+16+8hi .. +7]
//   C/D VGPR r -> row r + 8*(l>=16), col l&15
// ---------------------------------------------------------------------
__global__ __launch_bounds__(TPB) void gemm_f16(
    const _Float16* __restrict__ A, int lda,
    const _Float16* __restrict__ W, int ldw,
    const float* __restrict__ bias,
    float* __restrict__ C, int ldc,
    int M, int N, int Ka, int act)
{
  int wave   = (int)((blockIdx.x * blockDim.x + threadIdx.x) >> 5);
  int ntiles = (N + 31) >> 5;
  int mtiles = M >> 4;                       // M is a multiple of 16
  if (wave >= ntiles * mtiles) return;       // wave-uniform exit
  int tm = wave / ntiles, tn = wave - tm * ntiles;
  int row0 = tm << 4, col0 = tn << 5;
  int lane = (int)(threadIdx.x & 31);
  int m = lane & 15, hi = lane >> 4;

  const _Float16* Ar = A + (size_t)(row0 + m) * lda;
  const _Float16* Wr = W + col0 + m;

  v8f acc0 = {}, acc1 = {};
  for (int k0 = 0; k0 < Ka; k0 += 32) {
    V16 a;
    a.h[0] = *(const v8h*)(Ar + k0 + 8 * hi);
    a.h[1] = *(const v8h*)(Ar + k0 + 16 + 8 * hi);
    const _Float16* wp = Wr + (size_t)k0 * ldw;
    V16 b0, b1;
#pragma unroll
    for (int s = 0; s < 8; ++s) {
      b0.v[s]     = wp[(size_t)(8 * hi + s) * ldw];
      b0.v[8 + s] = wp[(size_t)(16 + 8 * hi + s) * ldw];
      b1.v[s]     = wp[(size_t)(8 * hi + s) * ldw + 16];
      b1.v[8 + s] = wp[(size_t)(16 + 8 * hi + s) * ldw + 16];
    }
    acc0 = __builtin_amdgcn_wmma_f32_16x16x32_f16(false, a.v, false, b0.v,
                                                  (short)0, acc0, false, false);
    acc1 = __builtin_amdgcn_wmma_f32_16x16x32_f16(false, a.v, false, b1.v,
                                                  (short)0, acc1, false, false);
  }

  int colA = col0 + m, colB = colA + 16;
  float bvA = (bias && colA < N) ? bias[colA] : 0.f;
  float bvB = (bias && colB < N) ? bias[colB] : 0.f;
#pragma unroll
  for (int r = 0; r < 8; ++r) {
    int row = row0 + r + (hi ? 8 : 0);
    if (colA < N) {
      float v = acc0[r] + bvA;
      if (act) v = elu1(v);
      C[(size_t)row * ldc + colA] = v;
    }
    if (colB < N) {
      float v = acc1[r] + bvB;
      if (act) v = elu1(v);
      C[(size_t)row * ldc + colB] = v;
    }
  }
}

// ---------------------------------------------------------------------
// Batched per-point fX[p,k,c] = sum_j X[p,k,j] * cat[p,j,c] on WMMA.
//   Xp:   f16 [BP, 16, 32], zeros where m>=K or j>=K
//   catp: f16 [BP, 16, Ccat] (Ccat multiple of 16), zeros where j>=K
// Lanes 16..31 load B rows j=8..15, B slots 8..15 (j>=16) are zeroed; the A-side zeros (j>=16)
// make their contribution 0, so no masking is needed.
// Output fXp is f16 [BP, K*Ccat] feeding the fused projection GEMM.
// ---------------------------------------------------------------------
__global__ __launch_bounds__(TPB) void fx_wmma(
    const _Float16* __restrict__ Xp,
    const _Float16* __restrict__ catp,
    _Float16* __restrict__ fXp,
    int BP, int K, int Ccat)
{
  int ctiles = Ccat >> 4;
  int wave = (int)((blockIdx.x * blockDim.x + threadIdx.x) >> 5);
  if (wave >= BP * ctiles) return;            // wave-uniform exit
  int p = wave / ctiles, ct = wave - p * ctiles;
  int lane = (int)(threadIdx.x & 31);
  int m = lane & 15, hi = lane >> 4;
  int c0 = ct << 4;

  const _Float16* xr = Xp + ((size_t)p * 16 + m) * 32;
  V16 a;
  a.h[0] = *(const v8h*)(xr + 8 * hi);
  a.h[1] = *(const v8h*)(xr + 16 + 8 * hi);

  const _Float16* cr = catp + (size_t)p * 16 * Ccat + c0 + m;
  V16 b;
#pragma unroll
  for (int s = 0; s < 8; ++s) {
    b.v[s] = cr[(size_t)(8 * hi + s) * Ccat];
    b.v[8 + s] = (_Float16)0;
  }

  v8f acc = {};
  acc = __builtin_amdgcn_wmma_f32_16x16x32_f16(false, a.v, false, b.v,
                                               (short)0, acc, false, false);
  int c = c0 + m;
#pragma unroll
  for (int r = 0; r < 8; ++r) {
    int k = r + (hi ? 8 : 0);
    if (k < K) fXp[((size_t)p * K + k) * Ccat + c] = (_Float16)acc[r];
  }
}

// ------------------------- pack kernels ------------------------------
// Generic f32 -> zero-padded f16: dst[r,c] (dstCols wide)
__global__ __launch_bounds__(TPB) void pack_f16(
    const float* __restrict__ src, int ld, int srcRows, int srcCols,
    _Float16* __restrict__ dst, int dstCols, long long total)
{
  long long t = (long long)blockIdx.x * blockDim.x + threadIdx.x;
  if (t >= total) return;
  int c = (int)(t % dstCols);
  long long r = t / dstCols;
  float v = (r < srcRows && c < srcCols) ? src[(size_t)r * ld + c] : 0.f;
  dst[t] = (_Float16)v;
}

// X [BP, K*K] f32 -> Xp [BP,16,32] f16 zero-padded
__global__ __launch_bounds__(TPB) void pack_x(
    const float* __restrict__ X, _Float16* __restrict__ Xp,
    long long total, int K, int KK)
{
  long long t = (long long)blockIdx.x * blockDim.x + threadIdx.x;
  if (t >= total) return;
  int j = (int)(t & 31);
  int m = (int)((t >> 5) & 15);
  long long bp = t >> 9;
  float v = (m < K && j < K) ? X[(size_t)bp * KK + m * K + j] : 0.f;
  Xp[t] = (_Float16)v;
}

// cat [BP*K, Ccat] f32 -> catp [BP,16,Ccat] f16, rows j>=K zero
__global__ __launch_bounds__(TPB) void pack_cat(
    const float* __restrict__ cat, _Float16* __restrict__ catp,
    long long total, int K, int Ccat)
{
  long long t = (long long)blockIdx.x * blockDim.x + threadIdx.x;
  if (t >= total) return;
  int c = (int)(t % Ccat);
  long long r = t / Ccat;
  int j = (int)(r & 15);
  long long bp = r >> 4;
  float v = (j < K) ? cat[((size_t)bp * K + j) * Ccat + c] : 0.f;
  catp[t] = (_Float16)v;
}

// ---------------------------------------------------------------------
// Brute-force kNN with dilation: keep D*K+1 nearest, drop self (slot 0),
// take slots 1, 1+D, ... ; q/ref are row-prefixes of x (one stride).
// ---------------------------------------------------------------------
__global__ __launch_bounds__(TPB) void knn_kernel(
    const float* __restrict__ pts, int ptsStride,
    int* __restrict__ idxOut,
    int B, int P, int Nref, int K, int D)
{
  int t = (int)(blockIdx.x * blockDim.x + threadIdx.x);
  if (t >= B * P) return;
  int b = t / P, p = t - b * P;
  const float* qp = pts + ((size_t)b * ptsStride + p) * 3;
  float qx = qp[0], qy = qp[1], qz = qp[2];

  float bd[MAX_DK1]; int bi[MAX_DK1];
  int DK1 = D * K + 1;
  for (int j = 0; j < DK1; ++j) { bd[j] = 3.4e38f; bi[j] = 0; }

  const float* rp = pts + (size_t)b * ptsStride * 3;
  for (int n = 0; n < Nref; ++n) {
    float dx = rp[n * 3 + 0] - qx;
    float dy = rp[n * 3 + 1] - qy;
    float dz = rp[n * 3 + 2] - qz;
    float d2 = dx * dx + dy * dy + dz * dz;
    if (d2 < bd[DK1 - 1]) {
      int j = DK1 - 1;
      while (j > 0 && bd[j - 1] > d2) { bd[j] = bd[j - 1]; bi[j] = bi[j - 1]; --j; }
      bd[j] = d2; bi[j] = n;
    }
  }
  for (int k = 0; k < K; ++k) idxOut[(size_t)t * K + k] = bi[1 + k * D];
}

// local[b,p,k,:] = pts[b, idx[b,p,k], :] - pts[b, p, :]
__global__ __launch_bounds__(TPB) void gather_local(
    const float* __restrict__ pts, int ptsStride,
    const int* __restrict__ idx, float* __restrict__ local,
    int total, int P, int K)
{
  int t = (int)(blockIdx.x * blockDim.x + threadIdx.x);
  if (t >= total) return;
  int bp = t / K;
  int b = bp / P, p = bp - b * P;
  int n = idx[t];
  const float* rp = pts + ((size_t)b * ptsStride + p) * 3;
  const float* np = pts + ((size_t)b * ptsStride + n) * 3;
  float* o = local + (size_t)t * 3;
  o[0] = np[0] - rp[0]; o[1] = np[1] - rp[1]; o[2] = np[2] - rp[2];
}

// dst[g, dstOff + c] = fts[b, idx[g], c]
__global__ __launch_bounds__(TPB) void gather_feat(
    const float* __restrict__ fts, int Nf, int C,
    const int* __restrict__ idx,
    float* __restrict__ dst, int dstStride, int dstOff,
    long long total, int PK)
{
  long long t = (long long)blockIdx.x * blockDim.x + threadIdx.x;
  if (t >= total) return;
  int c = (int)(t % C);
  long long g = t / C;
  int b = (int)(g / PK);
  int n = idx[g];
  dst[(size_t)g * dstStride + dstOff + c] = fts[((size_t)b * Nf + n) * C + c];
}

// dst[r, dstOff + c] = src[r, c]   (channel concat)
__global__ __launch_bounds__(TPB) void copy_cols(
    const float* __restrict__ src, int srcStride,
    float* __restrict__ dst, int dstStride, int dstOff,
    long long total, int C)
{
  long long t = (long long)blockIdx.x * blockDim.x + threadIdx.x;
  if (t >= total) return;
  int c = (int)(t % C);
  long long r = t / C;
  dst[(size_t)r * dstStride + dstOff + c] = src[(size_t)r * srcStride + c];
}

// Wf[k*Ccat + c, o] = sum_d dw[c,d,k] * pw[c*dm+d, o]  (emitted as f16)
__global__ __launch_bounds__(TPB) void wf_fuse(
    const float* __restrict__ dw,   // [Ccat, dm, K]
    const float* __restrict__ pw,   // [Ccat*dm, Cout]
    _Float16* __restrict__ Wf,      // [K*Ccat, Cout]
    long long total, int Ccat, int dm, int K, int Cout)
{
  long long t = (long long)blockIdx.x * blockDim.x + threadIdx.x;
  if (t >= total) return;
  int o = (int)(t % Cout);
  long long kc = t / Cout;
  int c = (int)(kc % Ccat);
  int k = (int)(kc / Ccat);
  float s = 0.f;
  for (int d = 0; d < dm; ++d)
    s += dw[((size_t)c * dm + d) * K + k] * pw[((size_t)c * dm + d) * Cout + o];
  Wf[(size_t)(k * Ccat + c) * Cout + o] = (_Float16)s;
}

// ============================ host side ==============================

struct Arena {
  char* base; size_t off;
  float*    f(size_t n) { char* p = base + off; off += ((n * 4 + 255) & ~(size_t)255); return (float*)p; }
  int*      i(size_t n) { char* p = base + off; off += ((n * 4 + 255) & ~(size_t)255); return (int*)p; }
  _Float16* h(size_t n) { char* p = base + off; off += ((n * 2 + 255) & ~(size_t)255); return (_Float16*)p; }
};

struct DenseP { const float *W, *b; };
struct XconvP { const float *d1W,*d1b,*d2W,*d2b,*dw,*pb,*pw,*x0W,*x0b,*x1W,*x1b,*x2W,*x2b; };

static inline unsigned blocks_for(long long n) { return (unsigned)((n + TPB - 1) / TPB); }
static inline int ru32(int x) { return (x + 31) & ~31; }

static void gemm_f16_launch(hipStream_t s, const _Float16* A, int lda,
                            const _Float16* W, int ldw, const float* bias,
                            float* C, int ldc, int M, int N, int Ka, int act)
{
  long long waves = (long long)(M / 16) * ((N + 31) / 32);
  gemm_f16<<<blocks_for(waves * 32), TPB, 0, s>>>(A, lda, W, ldw, bias, C, ldc, M, N, Ka, act);
}

// pack A (f32 [M,Kd], row-stride lda) and W (f32 [Kd,N]) to f16, then GEMM
static void gemm_pack(hipStream_t s, Arena& ar,
                      const float* A, int lda, int M, int Kd,
                      const float* W, int N, const float* bias,
                      float* C, int ldc, int act)
{
  int Ka = ru32(Kd), Np = ru32(N);
  _Float16* Ap = ar.h((size_t)M * Ka);
  pack_f16<<<blocks_for((long long)M * Ka), TPB, 0, s>>>(A, lda, M, Kd, Ap, Ka, (long long)M * Ka);
  _Float16* Wp = ar.h((size_t)Ka * Np);
  pack_f16<<<blocks_for((long long)Ka * Np), TPB, 0, s>>>(W, N, Kd, N, Wp, Np, (long long)Ka * Np);
  gemm_f16_launch(s, Ap, Ka, Wp, Np, bias, C, ldc, M, N, Ka, act);
}

// One full XConv block.
static void run_xconv(hipStream_t st, Arena& ar, const XconvP& p,
                      const float* x, int coordStride,
                      const float* fts, int Nref, int Cin,
                      int B, int P, int K, int D, int Cmid, int dm, int Cout,
                      float* out)
{
  int BP = B * P, BPK = BP * K, Ccat = Cmid + Cin, KK = K * K;

  int* idx = ar.i((size_t)BPK);
  knn_kernel<<<blocks_for(BP), TPB, 0, st>>>(x, coordStride, idx, B, P, Nref, K, D);

  float* local = ar.f((size_t)BPK * 3);
  gather_local<<<blocks_for(BPK), TPB, 0, st>>>(x, coordStride, idx, local, BPK, P, K);

  float* lift = ar.f((size_t)BPK * Cmid);
  float* cat  = ar.f((size_t)BPK * Ccat);
  gemm_pack(st, ar, local, 3,    BPK, 3,    p.d1W, Cmid, p.d1b, lift, Cmid, 1);
  gemm_pack(st, ar, lift,  Cmid, BPK, Cmid, p.d2W, Cmid, p.d2b, cat,  Ccat, 1);
  if (Cin > 0)
    gather_feat<<<blocks_for((long long)BPK * Cin), TPB, 0, st>>>(
        fts, Nref, Cin, idx, cat, Ccat, Cmid, (long long)BPK * Cin, P * K);

  // X-transform MLP on t = local viewed as [BP, 3K]
  float* xa = ar.f((size_t)BP * KK);
  float* xb = ar.f((size_t)BP * KK);
  gemm_pack(st, ar, local, 3 * K, BP, 3 * K, p.x0W, KK, p.x0b, xa, KK, 1);
  gemm_pack(st, ar, xa,    KK,    BP, KK,    p.x1W, KK, p.x1b, xb, KK, 1);
  gemm_pack(st, ar, xb,    KK,    BP, KK,    p.x2W, KK, p.x2b, xa, KK, 0);

  _Float16* Xp = ar.h((size_t)BP * 512);
  pack_x<<<blocks_for((long long)BP * 512), TPB, 0, st>>>(xa, Xp, (long long)BP * 512, K, KK);
  _Float16* catp = ar.h((size_t)BP * 16 * Ccat);
  pack_cat<<<blocks_for((long long)BP * 16 * Ccat), TPB, 0, st>>>(
      cat, catp, (long long)BP * 16 * Ccat, K, Ccat);

  _Float16* fXp = ar.h((size_t)BP * K * Ccat);
  {
    long long waves = (long long)BP * (Ccat / 16);
    fx_wmma<<<blocks_for(waves * 32), TPB, 0, st>>>(Xp, catp, fXp, BP, K, Ccat);
  }

  _Float16* Wfp = ar.h((size_t)K * Ccat * Cout);
  wf_fuse<<<blocks_for((long long)K * Ccat * Cout), TPB, 0, st>>>(
      p.dw, p.pw, Wfp, (long long)K * Ccat * Cout, Ccat, dm, K, Cout);

  // fused projection: K*Ccat and Cout are multiples of 32 for all layers
  gemm_f16_launch(st, fXp, K * Ccat, Wfp, Cout, p.pb, out, Cout, BP, Cout, K * Ccat, 1);
}

extern "C" void kernel_launch(void* const* d_in, const int* in_sizes, int n_in,
                              void* d_out, int out_size, void* d_ws, size_t ws_size,
                              hipStream_t stream)
{
  (void)in_sizes; (void)n_in; (void)out_size; (void)ws_size;
  // Input 0 is x [4,4096,3]; params follow as JAX-tree leaves (sorted keys):
  // dd0 dd1 dd2 deconv0 deconv1 deconv2 dense1 dense2 dense3 fc0 fc1 fc2
  // xconv1..4; xconv dict: d1.W d1.b d2.W d2.b dw pb pw x0.W x0.b x1.W x1.b x2.W x2.b
  const float* x = (const float*)d_in[0];
  int ii = 1;
  auto nd = [&](DenseP& d) { d.W = (const float*)d_in[ii]; d.b = (const float*)d_in[ii+1]; ii += 2; };
  auto nx = [&](XconvP& p) {
    p.d1W=(const float*)d_in[ii+0];  p.d1b=(const float*)d_in[ii+1];
    p.d2W=(const float*)d_in[ii+2];  p.d2b=(const float*)d_in[ii+3];
    p.dw =(const float*)d_in[ii+4];  p.pb =(const float*)d_in[ii+5];
    p.pw =(const float*)d_in[ii+6];
    p.x0W=(const float*)d_in[ii+7];  p.x0b=(const float*)d_in[ii+8];
    p.x1W=(const float*)d_in[ii+9];  p.x1b=(const float*)d_in[ii+10];
    p.x2W=(const float*)d_in[ii+11]; p.x2b=(const float*)d_in[ii+12];
    ii += 13;
  };
  DenseP dd0, dd1, dd2, dense1, dense2, dense3, fc0, fc1, fc2;
  XconvP dc0, dc1, dc2, xc1, xc2, xc3, xc4;
  nd(dd0); nd(dd1); nd(dd2);
  nx(dc0); nx(dc1); nx(dc2);
  nd(dense1); nd(dense2); nd(dense3);
  nd(fc0); nd(fc1); nd(fc2);
  nx(xc1); nx(xc2); nx(xc3); nx(xc4);

  const int B = 4, N1 = 4096, N2 = 1536, N3 = 768, N4 = 384, CS = 4096;

  Arena ar{ (char*)d_ws, 0 };
  // persistent activations (reps are all row-prefixes of x)
  float* out1 = ar.f((size_t)B * N1 * 64);
  float* fts1 = ar.f((size_t)B * N1 * 64);
  float* out2 = ar.f((size_t)B * N2 * 128);
  float* fts2 = ar.f((size_t)B * N2 * 128);
  float* out3 = ar.f((size_t)B * N3 * 256);
  float* fts3 = ar.f((size_t)B * N3 * 256);
  float* out4 = ar.f((size_t)B * N4 * 512);
  float* dbuf = ar.f((size_t)B * N3 * 256);
  float* d2b  = ar.f((size_t)B * N2 * 128);
  size_t mark = ar.off;   // stage scratch resets to here

  // ----- encoder -----
  run_xconv(stream, ar, xc1, x, CS, nullptr, N1, 0,   B, N1, 8,  1, 32,  4,  64,  out1);
  ar.off = mark;
  gemm_pack(stream, ar, out1, 64,  B * N1, 64,  dense1.W, 64,  dense1.b, fts1, 64,  1);
  ar.off = mark;

  run_xconv(stream, ar, xc2, x, CS, fts1, N1, 64,  B, N2, 12, 2, 32, 16, 128, out2);
  ar.off = mark;
  gemm_pack(stream, ar, out2, 128, B * N2, 128, dense2.W, 128, dense2.b, fts2, 128, 1);
  ar.off = mark;

  run_xconv(stream, ar, xc3, x, CS, fts2, N2, 128, B, N3, 16, 2, 64, 32, 256, out3);
  ar.off = mark;
  gemm_pack(stream, ar, out3, 256, B * N3, 256, dense3.W, 256, dense3.b, fts3, 256, 1);
  ar.off = mark;

  run_xconv(stream, ar, xc4, x, CS, fts3, N3, 256, B, N4, 16, 4, 128, 64, 512, out4);
  ar.off = mark;

  // ----- decoder -----
  {
    float* dt = ar.f((size_t)B * N3 * 256);
    run_xconv(stream, ar, dc0, x, CS, out4, N4, 512, B, N3, 16, 4, 128, 1, 256, dt);
    float* cc = ar.f((size_t)B * N3 * 512);
    copy_cols<<<blocks_for((long long)B * N3 * 256), TPB, 0, stream>>>(dt,   256, cc, 512, 0,   (long long)B * N3 * 256, 256);
    copy_cols<<<blocks_for((long long)B * N3 * 256), TPB, 0, stream>>>(out3, 256, cc, 512, 256, (long long)B * N3 * 256, 256);
    gemm_pack(stream, ar, cc, 512, B * N3, 512, dd0.W, 256, dd0.b, dbuf, 256, 1);
    ar.off = mark;
  }
  {
    float* dt = ar.f((size_t)B * N2 * 128);
    run_xconv(stream, ar, dc1, x, CS, dbuf, N3, 256, B, N2, 16, 2, 64, 1, 128, dt);
    float* cc = ar.f((size_t)B * N2 * 256);
    copy_cols<<<blocks_for((long long)B * N2 * 128), TPB, 0, stream>>>(dt,   128, cc, 256, 0,   (long long)B * N2 * 128, 128);
    copy_cols<<<blocks_for((long long)B * N2 * 128), TPB, 0, stream>>>(out2, 128, cc, 256, 128, (long long)B * N2 * 128, 128);
    gemm_pack(stream, ar, cc, 256, B * N2, 256, dd1.W, 128, dd1.b, d2b, 128, 1);
    ar.off = mark;
  }
  {
    float* dt = ar.f((size_t)B * N1 * 64);
    run_xconv(stream, ar, dc2, x, CS, d2b, N2, 128, B, N1, 12, 2, 32, 1, 64, dt);
    float* cc = ar.f((size_t)B * N1 * 128);
    copy_cols<<<blocks_for((long long)B * N1 * 64), TPB, 0, stream>>>(dt,   64, cc, 128, 0,  (long long)B * N1 * 64, 64);
    copy_cols<<<blocks_for((long long)B * N1 * 64), TPB, 0, stream>>>(out1, 64, cc, 128, 64, (long long)B * N1 * 64, 64);
    float* d3 = ar.f((size_t)B * N1 * 64);
    gemm_pack(stream, ar, cc, 128, B * N1, 128, dd2.W, 64, dd2.b, d3, 64, 1);
    float* t0 = ar.f((size_t)B * N1 * 64);
    float* t1 = ar.f((size_t)B * N1 * 64);
    gemm_pack(stream, ar, d3, 64, B * N1, 64, fc0.W, 64, fc0.b, t0, 64, 1);
    gemm_pack(stream, ar, t0, 64, B * N1, 64, fc1.W, 64, fc1.b, t1, 64, 1);
    // reference's dense() applies elu on the final layer too
    gemm_pack(stream, ar, t1, 64, B * N1, 64, fc2.W, 13, fc2.b, (float*)d_out, 13, 1);
    ar.off = mark;
  }
}
